// CircuitLossV2_34359738851
// MI455X (gfx1250) — matrix operand order repaired and checked by reading, not compiled, not run on hardware
//
#include <hip/hip_runtime.h>

typedef __attribute__((ext_vector_type(16))) _Float16 v16h;
typedef __attribute__((ext_vector_type(8)))  float    v8f;

#define BB 64
#define TT 1024
#define NTOK (BB*TT)
#define NCH 32            // K chunks of 32 tokens
#define CH_HALVES 8192    // 256 * 32 halves per chunk

// ws layout (bytes):
//  0        : float scalars[16] (0:type 1:na 2:nb 3:val 4:self 5:mask 6:dup)
//  64       : float bacc[64*4]
//  4096     : _Float16 Aw[64*1024*256]  K-tiled [b][c][i][kk]   32 MB
//  4096+32M : _Float16 Bw[...]          same layout             32 MB
//  4096+64M : float ec[64*256*256]                              16 MB

__device__ __forceinline__ unsigned lds_addr32(const void* p) {
    // generic LDS address = {SHARED_BASE[63:32], lds_offset[31:0]}
    return (unsigned)(unsigned long long)p;
}

__device__ __forceinline__ void async_copy_b128(unsigned lds_off, const void* gptr) {
    asm volatile("global_load_async_to_lds_b128 %0, %1, off"
                 :: "v"(lds_off), "v"((unsigned long long)gptr) : "memory");
}

#if __has_builtin(__builtin_amdgcn_s_wait_asynccnt)
#define WAIT_ASYNC(n) __builtin_amdgcn_s_wait_asynccnt(n)
#else
#define WAIT_ASYNC(n) asm volatile("s_wait_asynccnt %0" :: "i"(n))
#endif

__global__ __launch_bounds__(512) void k_zero(float* scalars, float* bacc) {
    int t = threadIdx.x;
    if (t < 16) scalars[t] = 0.0f;
    int u = t - 16;
    if (u >= 0 && u < 256) bacc[u] = 0.0f;
}

__global__ __launch_bounds__(256) void k_pointwise(
    const float* __restrict__ type_logits,
    const float* __restrict__ na_logits,
    const float* __restrict__ nb_logits,
    const float* __restrict__ values,
    const float* __restrict__ sequence,
    _Float16* __restrict__ Aw, _Float16* __restrict__ Bw,
    float* __restrict__ scalars, float* __restrict__ bacc)
{
    __shared__ _Float16 TA[256 * 9];   // [i][wid], padded stride 9
    __shared__ _Float16 TB[256 * 9];
    __shared__ float red[8][12];

    const int lane = threadIdx.x & 31;
    const int wid  = threadIdx.x >> 5;
    const int tok  = blockIdx.x * 8 + wid;
    const int t    = tok & (TT - 1);

    // shifted target (next token), zeros at sequence end
    float s0 = 0.f, s1 = 0.f, s2 = 0.f, s3 = 0.f;
    if (t < TT - 1) {
        const float* sq = sequence + (size_t)(tok + 1) * 4;
        s0 = sq[0]; s1 = sq[1]; s2 = sq[2]; s3 = sq[3];
    }
    const int t_type = (int)s0;
    const int t_a    = (int)s1;
    const int t_b    = (int)s2;
    const float t_val = s3;
    const float mask = (t_type >= 3 && t_type <= 5) ? 1.0f : 0.0f;

    // ---- type softmax over 16 (duplicated across lane halves) ----
    const float* tl = type_logits + (size_t)tok * 16;
    float x = tl[lane & 15];
    float m = x;
    #pragma unroll
    for (int off = 8; off; off >>= 1) m = fmaxf(m, __shfl_xor(m, off));
    float se = __expf(x - m);
    #pragma unroll
    for (int off = 8; off; off >>= 1) se += __shfl_xor(se, off);
    const float lseT = m + __logf(se);
    const float typeCE = lseT - __shfl(x, t_type);
    const float p_comp = __expf(__shfl(x, 3) - lseT)
                       + __expf(__shfl(x, 4) - lseT)
                       + __expf(__shfl(x, 5) - lseT);

    // ---- node_a softmax over 256 (lane owns 8 contiguous) ----
    const float4* a4 = (const float4*)(na_logits + (size_t)tok * 256 + lane * 8);
    float4 va0 = a4[0], va1 = a4[1];
    float av[8] = {va0.x, va0.y, va0.z, va0.w, va1.x, va1.y, va1.z, va1.w};
    float ma = av[0];
    #pragma unroll
    for (int j = 1; j < 8; ++j) ma = fmaxf(ma, av[j]);
    #pragma unroll
    for (int off = 16; off; off >>= 1) ma = fmaxf(ma, __shfl_xor(ma, off));
    float sa = 0.f;
    #pragma unroll
    for (int j = 0; j < 8; ++j) sa += __expf(av[j] - ma);
    #pragma unroll
    for (int off = 16; off; off >>= 1) sa += __shfl_xor(sa, off);
    const float lseA = ma + __logf(sa);
    float pa[8];
    #pragma unroll
    for (int j = 0; j < 8; ++j) pa[j] = __expf(av[j] - lseA);
    float cand = 0.f;
    #pragma unroll
    for (int j = 0; j < 8; ++j) cand = ((t_a & 7) == j) ? av[j] : cand;
    const float ceA = lseA - __shfl(cand, t_a >> 3);

    // ---- node_b softmax over 256 ----
    const float4* b4 = (const float4*)(nb_logits + (size_t)tok * 256 + lane * 8);
    float4 vb0 = b4[0], vb1 = b4[1];
    float bv[8] = {vb0.x, vb0.y, vb0.z, vb0.w, vb1.x, vb1.y, vb1.z, vb1.w};
    float mb = bv[0];
    #pragma unroll
    for (int j = 1; j < 8; ++j) mb = fmaxf(mb, bv[j]);
    #pragma unroll
    for (int off = 16; off; off >>= 1) mb = fmaxf(mb, __shfl_xor(mb, off));
    float sb = 0.f;
    #pragma unroll
    for (int j = 0; j < 8; ++j) sb += __expf(bv[j] - mb);
    #pragma unroll
    for (int off = 16; off; off >>= 1) sb += __shfl_xor(sb, off);
    const float lseB = mb + __logf(sb);
    float pb[8];
    #pragma unroll
    for (int j = 0; j < 8; ++j) pb[j] = __expf(bv[j] - lseB);
    cand = 0.f;
    #pragma unroll
    for (int j = 0; j < 8; ++j) cand = ((t_b & 7) == j) ? bv[j] : cand;
    const float ceB = lseB - __shfl(cand, t_b >> 3);

    // ---- selfloop, value ----
    float sl = 0.f;
    #pragma unroll
    for (int j = 0; j < 8; ++j) sl += pa[j] * pb[j];
    #pragma unroll
    for (int off = 16; off; off >>= 1) sl += __shfl_xor(sl, off);
    const float pv = values[tok];
    const float vloss = (pv - t_val) * (pv - t_val) * mask;

    // gnd/in terms: indices 0,1 live on lane 0 (j = 0,1)
    const float pa0 = __shfl(pa[0], 0), pa1 = __shfl(pa[1], 0);
    const float pb0 = __shfl(pb[0], 0), pb1 = __shfl(pb[1], 0);

    // ---- transpose f16 operands through LDS: column = wid ----
    #pragma unroll
    for (int j = 0; j < 8; ++j) {
        const int i = lane * 8 + j;
        TA[i * 9 + wid] = (_Float16)(mask * pa[j]);
        TB[i * 9 + wid] = (_Float16)pb[j];
    }
    if (lane == 0) {
        red[wid][0] = typeCE;
        red[wid][1] = ceA * mask;
        red[wid][2] = ceB * mask;
        red[wid][3] = vloss;
        red[wid][4] = sl * mask;
        red[wid][5] = mask;
        red[wid][6] = pa0 * p_comp;
        red[wid][7] = pb0 * p_comp;
        red[wid][8] = pa1 * p_comp;
        red[wid][9] = pb1 * p_comp;
    }
    __syncthreads();

    // ---- coalesced K-tiled store: Aw[b][t/32][i][t%32] ----
    {
        const int i = threadIdx.x;            // 0..255
        union { _Float16 h[8]; uint4 q; } oa, ob;
        #pragma unroll
        for (int j = 0; j < 8; ++j) { oa.h[j] = TA[i * 9 + j]; ob.h[j] = TB[i * 9 + j]; }
        const int bb = (int)((blockIdx.x * 8) >> 10);
        const int t0 = (blockIdx.x * 8) & (TT - 1);
        const size_t cb = (size_t)bb * (TT * 256) + (size_t)(t0 >> 5) * CH_HALVES;
        *(uint4*)(Aw + cb + (size_t)i * 32 + (t0 & 31)) = oa.q;
        *(uint4*)(Bw + cb + (size_t)i * 32 + (t0 & 31)) = ob.q;
    }

    // ---- block reduction of 10 partial sums ----
    const int tid = threadIdx.x;
    if (tid < 10) {
        float acc = 0.f;
        #pragma unroll
        for (int w = 0; w < 8; ++w) acc += red[w][tid];
        if (tid < 6) {
            atomicAdd(&scalars[tid], acc);
        } else {
            const int bb = (int)((blockIdx.x * 8) >> 10);
            atomicAdd(&bacc[bb * 4 + (tid - 6)], acc);
        }
    }
}

// ec[b] (256x256 f32) = A[b]^T B[b] over K=1024, f16 WMMA.
// One workgroup = 64x64 output tile; async-to-LDS double-buffered staging.
__global__ __launch_bounds__(128) void k_gemm(
    const _Float16* __restrict__ Aw, const _Float16* __restrict__ Bw,
    float* __restrict__ ec)
{
    __shared__ unsigned short As[2][64 * 40];  // [i_loc][k], 80B padded rows
    __shared__ unsigned short Bs[2][64 * 40];  // [j_loc][k]

    const int b  = blockIdx.x >> 4;
    const int tb = blockIdx.x & 15;
    const int mb = (tb >> 2) << 6;
    const int nb = (tb & 3) << 6;
    const int tid  = threadIdx.x;
    const int lane = tid & 31;
    const int w    = tid >> 5;
    const int lm   = lane & 15;
    const int lh   = lane >> 4;

    const unsigned short* Ag = (const unsigned short*)Aw + (size_t)b * (TT * 256);
    const unsigned short* Bg = (const unsigned short*)Bw + (size_t)b * (TT * 256);

    // stage chunk c (64 rows x 32 halves, contiguous 64B rows) into buffer p
    auto stage = [&](int c, int p) {
        const size_t cbase = (size_t)c * CH_HALVES;
        #pragma unroll
        for (int it = 0; it < 2; ++it) {
            const int linear = it * 128 + tid;  // 0..255
            const int row = linear >> 2;        // 0..63
            const int qq  = linear & 3;         // 16B quad within row
            async_copy_b128(lds_addr32(&As[p][row * 40 + qq * 8]),
                            Ag + cbase + (size_t)(mb + row) * 32 + qq * 8);
            async_copy_b128(lds_addr32(&Bs[p][row * 40 + qq * 8]),
                            Bg + cbase + (size_t)(nb + row) * 32 + qq * 8);
        }
    };

    v8f acc[4] = {};
    stage(0, 0);

    for (int c = 0; c < NCH; ++c) {
        const int p = c & 1;
        if (c + 1 < NCH) { stage(c + 1, p ^ 1); WAIT_ASYNC(4); }
        else             { WAIT_ASYNC(0); }
        __syncthreads();

        // A fragment (16x32 f16): lane m = w*16+lm, VGPR r holds K pair
        union { v16h v; unsigned int u[8]; } fa;
        const int mloc = w * 16 + lm;
        #pragma unroll
        for (int r = 0; r < 8; ++r) {
            const int k0 = (r < 4 ? 2 * r : 16 + 2 * (r - 4)) + 8 * lh;
            fa.u[r] = *(const unsigned int*)(&As[p][mloc * 40 + k0]);
        }
        #pragma unroll
        for (int nj = 0; nj < 4; ++nj) {
            union { v16h v; unsigned int u[8]; } fb;
            const int nloc = nj * 16 + lm;
            #pragma unroll
            for (int r = 0; r < 8; ++r) {
                const int k0 = (r < 4 ? 2 * r : 16 + 2 * (r - 4)) + 8 * lh;
                fb.u[r] = *(const unsigned int*)(&Bs[p][nloc * 40 + k0]);
            }
            acc[nj] = __builtin_amdgcn_wmma_f32_16x16x32_f16(
                false, fa.v, false, fb.v, (short)0, acc[nj], false, false);
        }
        __syncthreads();
    }

    // D vgpr r, lane l -> row r + 8*(l>>4), col l&15
    float* eb = ec + ((size_t)b << 16);
    #pragma unroll
    for (int nj = 0; nj < 4; ++nj) {
        #pragma unroll
        for (int r = 0; r < 8; ++r) {
            const int ig = mb + w * 16 + r + 8 * lh;
            const int jg = nb + nj * 16 + lm;
            eb[(size_t)ig * 256 + jg] = acc[nj][r];
        }
    }
}

__global__ __launch_bounds__(256) void k_dup(const float* __restrict__ ec,
                                             float* __restrict__ scalars)
{
    __shared__ float red[256];
    float acc = 0.f;
    const size_t total = (size_t)BB * 65536;
    for (size_t idx = (size_t)blockIdx.x * 256 + threadIdx.x; idx < total;
         idx += (size_t)gridDim.x * 256) {
        const int b = (int)(idx >> 16);
        const int r = (int)(idx & 65535);
        const int i = r >> 8;
        const int j = r & 255;
        const float e  = ec[idx];
        const float et = ec[((size_t)b << 16) + (size_t)j * 256 + i];
        const float s = e + et - 1.0f;
        acc += (s > 0.f) ? s * s : 0.f;
    }
    red[threadIdx.x] = acc;
    __syncthreads();
    #pragma unroll
    for (int off = 128; off; off >>= 1) {
        if (threadIdx.x < off) red[threadIdx.x] += red[threadIdx.x + off];
        __syncthreads();
    }
    if (threadIdx.x == 0) atomicAdd(&scalars[6], red[0]);
}

__global__ void k_final(const float* __restrict__ scalars,
                        const float* __restrict__ bacc,
                        float* __restrict__ out)
{
    if (threadIdx.x == 0 && blockIdx.x == 0) {
        const float denom = scalars[5] + 1e-8f;
        const float type_loss  = scalars[0] * (1.0f / (float)NTOK);
        const float node_loss  = (scalars[1] + scalars[2]) * 0.5f / denom;
        const float value_loss = scalars[3] / denom;
        const float selfp      = scalars[4] / denom;
        const float dup        = scalars[6] * (1.0f / ((float)BB * 65536.0f));
        float g = 0.f, gi = 0.f;
        for (int b = 0; b < BB; ++b) {
            g  += __expf(-(bacc[b * 4 + 0] + bacc[b * 4 + 1]));
            gi += __expf(-(bacc[b * 4 + 2] + bacc[b * 4 + 3]));
        }
        const float gnd = (g + gi) * (1.0f / (float)BB);
        out[0] = 1.0f * type_loss + 0.5f * node_loss + 1.0f * value_loss
               + 2.0f * selfp + 1.0f * dup + 0.5f * gnd;
    }
}

extern "C" void kernel_launch(void* const* d_in, const int* in_sizes, int n_in,
                              void* d_out, int out_size, void* d_ws, size_t ws_size,
                              hipStream_t stream)
{
    const float* type_logits = (const float*)d_in[0];
    const float* na_logits   = (const float*)d_in[1];
    const float* nb_logits   = (const float*)d_in[2];
    const float* values      = (const float*)d_in[3];
    const float* sequence    = (const float*)d_in[4];

    char* ws = (char*)d_ws;
    float*    scalars = (float*)ws;
    float*    bacc    = (float*)(ws + 64);
    _Float16* Aw      = (_Float16*)(ws + 4096);
    _Float16* Bw      = (_Float16*)(ws + 4096 + 33554432ull);
    float*    ec      = (float*)(ws + 4096 + 67108864ull);

    k_zero     <<<1, 512, 0, stream>>>(scalars, bacc);
    k_pointwise<<<NTOK / 8, 256, 0, stream>>>(type_logits, na_logits, nb_logits,
                                              values, sequence, Aw, Bw, scalars, bacc);
    k_gemm     <<<BB * 16, 128, 0, stream>>>(Aw, Bw, ec);
    k_dup      <<<512, 256, 0, stream>>>(ec, scalars);
    k_final    <<<1, 64, 0, stream>>>(scalars, bacc, (float*)d_out);
}
